// NSDeltaNetLayer_62302795596470
// MI455X (gfx1250) — matrix operand order repaired and checked by reading, not compiled, not run on hardware
//
#include <hip/hip_runtime.h>
#include <math.h>

typedef __attribute__((ext_vector_type(16))) __bf16 v16bf;
typedef __attribute__((ext_vector_type(8)))  __bf16 v8bf;
typedef __attribute__((ext_vector_type(8)))  float  v8f;

#define BKK  64   // K elements staged per pipeline stage (2 WMMA K-steps)
#define APAD 8    // row stride 72 bf16 = 144B -> conflict-free b128 frag loads

// ---------------------------------------------------------------------------
// async global->LDS copies (CDNA5 GLOBAL_LOAD_ASYNC_TO_LDS_*), ASYNCcnt-tracked
// lds operand = low 32 bits of the generic address (LDS aperture keeps the
// byte offset in addr[31:0]).
// ---------------------------------------------------------------------------
__device__ __forceinline__ void cp_async_b128(void* lds, const void* g) {
    const unsigned lds_addr = (unsigned)(size_t)lds;
    asm volatile("global_load_async_to_lds_b128 %0, %1, off"
                 :: "v"(lds_addr), "v"(g) : "memory");
}
__device__ __forceinline__ void cp_async_b32(void* lds, const void* g) {
    const unsigned lds_addr = (unsigned)(size_t)lds;
    asm volatile("global_load_async_to_lds_b32 %0, %1, off"
                 :: "v"(lds_addr), "v"(g) : "memory");
}
__device__ __forceinline__ void wait_async_6()  { asm volatile("s_wait_asynccnt 0x6"  ::: "memory"); }
__device__ __forceinline__ void wait_async_25() { asm volatile("s_wait_asynccnt 0x19" ::: "memory"); }
__device__ __forceinline__ void wait_async_0()  { asm volatile("s_wait_asynccnt 0x0"  ::: "memory"); }

// ---------------------------------------------------------------------------
// C[M,N] = act( A[M,K] * W[N,K]^T ), bf16 in, f32 accumulate via WMMA.
// 256 threads (8 waves), block tile 64(M) x 128(N), 64-K stages, double-
// buffered LDS fed by the async copy engine one stage ahead of the WMMAs.
// ---------------------------------------------------------------------------
template <bool SILU, bool OUTBF>
__global__ __launch_bounds__(256) void gemm_bf16_wmma(const __bf16* __restrict__ A,
                                                      const __bf16* __restrict__ W,
                                                      void* __restrict__ Cvoid,
                                                      int M, int N, int K) {
    __shared__ __align__(16) __bf16 sA[2][64][BKK + APAD];
    __shared__ __align__(16) __bf16 sB[2][128][BKK + APAD];

    const int tid   = threadIdx.x;
    const int wave  = tid >> 5;
    const int lane  = tid & 31;
    const int half  = lane >> 4;
    const int l16   = lane & 15;
    const int wm    = (wave & 1) * 32;
    const int wn    = (wave >> 1) * 32;
    const int mBase = blockIdx.y * 64;
    const int nBase = blockIdx.x * 128;

    v8f acc[2][2];
#pragma unroll
    for (int mi = 0; mi < 2; ++mi)
#pragma unroll
        for (int ni = 0; ni < 2; ++ni)
            acc[mi][ni] = (v8f){0.f, 0.f, 0.f, 0.f, 0.f, 0.f, 0.f, 0.f};

    auto stage = [&](int buf, int k0) {
        // A: 64 rows x 128B = 512 16B-chunks; 2 per thread
#pragma unroll
        for (int it = 0; it < 2; ++it) {
            const int c = tid + 256 * it, r = c >> 3, col = (c & 7) * 8;
            cp_async_b128(&sA[buf][r][col], A + (size_t)(mBase + r) * K + k0 + col);
        }
        // B: 128 rows x 128B = 1024 chunks; 4 per thread
#pragma unroll
        for (int it = 0; it < 4; ++it) {
            const int c = tid + 256 * it, r = c >> 3, col = (c & 7) * 8;
            cp_async_b128(&sB[buf][r][col], W + (size_t)(nBase + r) * K + k0 + col);
        }
    };

    auto compute = [&](int cur) {
#pragma unroll
        for (int ks = 0; ks < BKK; ks += 32) {
            v16bf afr[2], bfr[2];
#pragma unroll
            for (int mi = 0; mi < 2; ++mi) {
                const int am = wm + mi * 16 + l16;
                const int ak = ks + half * 8;
                v8bf lo = *(const v8bf*)&sA[cur][am][ak];
                v8bf hi = *(const v8bf*)&sA[cur][am][ak + 16];
                afr[mi] = __builtin_shufflevector(lo, hi, 0, 1, 2, 3, 4, 5, 6, 7,
                                                  8, 9, 10, 11, 12, 13, 14, 15);
            }
#pragma unroll
            for (int ni = 0; ni < 2; ++ni) {
                const int bn = wn + ni * 16 + l16;
                const int bk = ks + half * 8;
                v8bf lo = *(const v8bf*)&sB[cur][bn][bk];
                v8bf hi = *(const v8bf*)&sB[cur][bn][bk + 16];
                bfr[ni] = __builtin_shufflevector(lo, hi, 0, 1, 2, 3, 4, 5, 6, 7,
                                                  8, 9, 10, 11, 12, 13, 14, 15);
            }
#pragma unroll
            for (int mi = 0; mi < 2; ++mi)
#pragma unroll
                for (int ni = 0; ni < 2; ++ni)
                    acc[mi][ni] = __builtin_amdgcn_wmma_f32_16x16x32_bf16(
                        false, afr[mi], false, bfr[ni], (short)0, acc[mi][ni], false, false);
        }
    };

    const int nT = K / BKK;
    stage(0, 0);
    for (int kt = 0; kt < nT - 1; ++kt) {          // branch-free steady state
        stage((kt + 1) & 1, (kt + 1) * BKK);
        wait_async_6();                             // oldest 6 (current stage) landed
        __syncthreads();
        compute(kt & 1);
        __syncthreads();                            // reads done before refill
    }
    wait_async_0();
    __syncthreads();
    compute((nT - 1) & 1);

    // epilogue: lane l16 = col, VGPR r = row (r + 8*half)
#pragma unroll
    for (int mi = 0; mi < 2; ++mi)
#pragma unroll
        for (int ni = 0; ni < 2; ++ni) {
            const int col = nBase + wn + ni * 16 + l16;
#pragma unroll
            for (int r = 0; r < 8; ++r) {
                const int row = mBase + wm + mi * 16 + half * 8 + r;
                float v = acc[mi][ni][r];
                if (SILU) v = v / (1.f + __expf(-v));
                if (OUTBF) ((__bf16*)Cvoid)[(size_t)row * N + col] = (__bf16)v;
                else       ((float*)Cvoid)[(size_t)row * N + col] = v;
            }
        }
}

// ---------------------------------------------------------------------------
// f32 -> bf16 bulk convert, 8 elements / thread
// ---------------------------------------------------------------------------
__global__ __launch_bounds__(256) void f32_to_bf16_kernel(const float* __restrict__ s,
                                                          __bf16* __restrict__ d) {
    const size_t i = (size_t)(blockIdx.x * blockDim.x + threadIdx.x) * 8;
    const float4 a = *(const float4*)(s + i);
    const float4 b = *(const float4*)(s + i + 4);
    v8bf o;
    o[0] = (__bf16)a.x; o[1] = (__bf16)a.y; o[2] = (__bf16)a.z; o[3] = (__bf16)a.w;
    o[4] = (__bf16)b.x; o[5] = (__bf16)b.y; o[6] = (__bf16)b.z; o[7] = (__bf16)b.w;
    *(v8bf*)(d + i) = o;
}

// ---------------------------------------------------------------------------
// beta = 2*sigmoid(x @ Wbeta^T)
// ---------------------------------------------------------------------------
__global__ void beta_kernel(const float* __restrict__ x, const float* __restrict__ Wb,
                            float* __restrict__ beta, int Kdim, int Nb) {
    const int idx = blockIdx.x * blockDim.x + threadIdx.x;
    const int row = idx / Nb, j = idx % Nb;
    const float4* xr = (const float4*)(x + (size_t)row * Kdim);
    const float4* wr = (const float4*)(Wb + (size_t)j * Kdim);
    float s = 0.f;
    for (int kk = 0; kk < Kdim / 4; ++kk) {
        float4 a = xr[kk], b = wr[kk];
        s += a.x * b.x + a.y * b.y + a.z * b.z + a.w * b.w;
    }
    beta[idx] = 2.f / (1.f + __expf(-s));
}

// ---------------------------------------------------------------------------
// delta-rule recurrence: ONE wave32 per (batch,head).
//  - single-wave workgroup: barriers are HW NOPs, same-wave LDS ops in-order
//  - q/k/v/beta async-staged 32 timesteps ahead (double-buffered)
//  - L2-norm scales factored out of the matvecs (wave-uniform)
//  - lane L owns rows/cols 2L,2L+1 of the 64x64 state M (stride 66 floats:
//    conflict-free b64 column reads and b64 row reads/updates)
// ---------------------------------------------------------------------------
#define TPRE 32
__global__ __launch_bounds__(32) void delta_rec(const __bf16* __restrict__ q,
                                                const __bf16* __restrict__ k,
                                                const __bf16* __restrict__ v,
                                                const float* __restrict__ beta,
                                                float* __restrict__ out,
                                                int S_, int Nh, int Hd, int Dd) {
    const int b = blockIdx.x / Nh, n = blockIdx.x % Nh;
    const int L  = threadIdx.x;        // 0..31
    const int h0 = 2 * L, h1 = h0 + 1;

    __shared__ __align__(16) float  Mst[64][66];
    __shared__ __align__(16) __bf16 qt[2][TPRE][64];
    __shared__ __align__(16) __bf16 kt[2][TPRE][64];
    __shared__ __align__(16) __bf16 vt[2][TPRE][64];
    __shared__ __align__(16) float  btf[2][TPRE];
    __shared__ __align__(16) float  dl[64];

#pragma unroll 4
    for (int i = 0; i < 64; ++i) { Mst[h0][i] = 0.f; Mst[h1][i] = 0.f; }

    const size_t headOff = (size_t)n * Hd;

    auto stageT = [&](int buf, int t0) {
#pragma unroll
        for (int it = 0; it < 8; ++it) {
            const int c = L + 32 * it;           // 0..255 chunks of 16B
            const int r = c >> 3, col = (c & 7) * 8;
            const size_t g = (size_t)(b * S_ + t0 + r) * Dd + headOff + col;
            cp_async_b128(&qt[buf][r][col], q + g);
            cp_async_b128(&kt[buf][r][col], k + g);
            cp_async_b128(&vt[buf][r][col], v + g);
        }
        cp_async_b32(&btf[buf][L], beta + (size_t)(b * S_ + t0 + L) * Nh + n);
    };

    stageT(0, 0);
    const int nTiles = S_ / TPRE;
    for (int tt = 0; tt < nTiles; ++tt) {
        const int buf = tt & 1;
        __syncthreads();                       // drain reads of buf^1 (HW NOP barrier)
        if (tt + 1 < nTiles) { stageT(buf ^ 1, (tt + 1) * TPRE); wait_async_25(); }
        else                 { wait_async_0(); }
        __syncthreads();

        for (int j = 0; j < TPRE; ++j) {
            const float qh0 = (float)qt[buf][j][h0], qh1 = (float)qt[buf][j][h1];
            const float kh0 = (float)kt[buf][j][h0], kh1 = (float)kt[buf][j][h1];
            float sq = qh0 * qh0 + qh1 * qh1;
            float sk = kh0 * kh0 + kh1 * kh1;
#pragma unroll
            for (int m = 16; m > 0; m >>= 1) {
                sq += __shfl_xor(sq, m, 32);
                sk += __shfl_xor(sk, m, 32);
            }
            const float qsc = 1.f / (sqrtf(sq) + 1e-6f);
            const float ksc = 1.f / (sqrtf(sk) + 1e-6f);

            float o0 = 0.f, o1 = 0.f, r0 = 0.f, r1 = 0.f;
#pragma unroll 2
            for (int i0 = 0; i0 < 64; i0 += 8) {
                const v8bf qv = *(const v8bf*)&qt[buf][j][i0];   // broadcast
                const v8bf kv = *(const v8bf*)&kt[buf][j][i0];   // broadcast
#pragma unroll
                for (int u = 0; u < 4; ++u) {
                    const int i = i0 + 2 * u;
                    const float2 c0 = *(const float2*)&Mst[i][h0];     // column pair
                    const float2 c1 = *(const float2*)&Mst[i + 1][h0];
                    const float2 ra = *(const float2*)&Mst[h0][i];     // row pairs
                    const float2 rb = *(const float2*)&Mst[h1][i];
                    const float q0 = (float)qv[2 * u], q1 = (float)qv[2 * u + 1];
                    const float k0 = (float)kv[2 * u], k1 = (float)kv[2 * u + 1];
                    o0 += q0 * c0.x + q1 * c1.x;
                    o1 += q0 * c0.y + q1 * c1.y;
                    r0 += ra.x * k0 + ra.y * k1;
                    r1 += rb.x * k0 + rb.y * k1;
                }
            }
            // delta (value axis), shared for the rank-1 update
            const float2 dv2 = { (float)vt[buf][j][h0] - ksc * r0,
                                 (float)vt[buf][j][h1] - ksc * r1 };
            *(float2*)&dl[h0] = dv2;

            // out (uses pre-update M)
            const float2 ov = { qsc * o0, qsc * o1 };
            *(float2*)(out + (size_t)(b * S_ + tt * TPRE + j) * Dd + headOff + h0) = ov;

            // rank-1 update of rows h0,h1 (same-wave LDS ops are in-order)
            const float bt  = btf[buf][j];
            const float bk0 = bt * ksc * kh0, bk1 = bt * ksc * kh1;
#pragma unroll 4
            for (int i = 0; i < 64; i += 2) {
                const float2 dd = *(const float2*)&dl[i];          // broadcast
                float2 a = *(const float2*)&Mst[h0][i];
                a.x += bk0 * dd.x; a.y += bk0 * dd.y;
                *(float2*)&Mst[h0][i] = a;
                float2 c = *(const float2*)&Mst[h1][i];
                c.x += bk1 * dd.x; c.y += bk1 * dd.y;
                *(float2*)&Mst[h1][i] = c;
            }
        }
    }
}

// ---------------------------------------------------------------------------
// RMSNorm over D per row, f32 in -> bf16 out (feeds final WMMA GEMM)
// ---------------------------------------------------------------------------
__global__ __launch_bounds__(256) void rmsnorm_bf16_kernel(const float* __restrict__ a,
                                                           const float* __restrict__ w,
                                                           __bf16* __restrict__ o, int Dd) {
    __shared__ float red[256];
    const int tid = threadIdx.x;
    const float* r = a + (size_t)blockIdx.x * Dd;
    __bf16* ro = o + (size_t)blockIdx.x * Dd;
    float s = 0.f;
    for (int i = tid; i < Dd; i += 256) { const float t = r[i]; s += t * t; }
    red[tid] = s;
    __syncthreads();
    for (int off = 128; off > 0; off >>= 1) {
        if (tid < off) red[tid] += red[tid + off];
        __syncthreads();
    }
    const float inv = 1.f / sqrtf(red[0] / (float)Dd + 1e-6f);
    for (int i = tid; i < Dd; i += 256) ro[i] = (__bf16)(r[i] * inv * w[i]);
}

// ---------------------------------------------------------------------------
extern "C" void kernel_launch(void* const* d_in, const int* in_sizes, int n_in,
                              void* d_out, int out_size, void* d_ws, size_t ws_size,
                              hipStream_t stream) {
    (void)in_sizes; (void)n_in; (void)out_size; (void)ws_size;
    const float* x  = (const float*)d_in[0];
    const float* Wq = (const float*)d_in[1];
    const float* Wk = (const float*)d_in[2];
    const float* Wv = (const float*)d_in[3];
    const float* Wo = (const float*)d_in[4];
    const float* Wb = (const float*)d_in[5];
    const float* nw = (const float*)d_in[6];

    const int Bb = 4, Ss = 4096, Dd = 1024, Nh = 16, Hd = 64;
    const int Mrows = Bb * Ss;                 // 16384
    const size_t tile = (size_t)Mrows * Dd;    // 16M elements
    const size_t wsz  = (size_t)Dd * Dd;       // 1M elements

    char* p = (char*)d_ws;
    __bf16* xbf  = (__bf16*)p;             p += tile * 2;
    __bf16* wqbf = (__bf16*)p;             p += wsz * 2;
    __bf16* wkbf = (__bf16*)p;             p += wsz * 2;
    __bf16* wvbf = (__bf16*)p;             p += wsz * 2;
    __bf16* wobf = (__bf16*)p;             p += wsz * 2;
    __bf16* qbf  = (__bf16*)p;             p += tile * 2;
    __bf16* kbf  = (__bf16*)p;             p += tile * 2;
    __bf16* vbf  = (__bf16*)p;             p += tile * 2;
    float*  attn = (float*)p;              p += tile * 4;
    __bf16* atbf = (__bf16*)p;             p += tile * 2;
    float*  bb   = (float*)p;              /* Mrows*Nh f32 */

    dim3 blk(256);
    dim3 grd(Dd / 128, Mrows / 64);  // (8, 256)

    f32_to_bf16_kernel<<<(int)(tile / (8 * 256)), 256, 0, stream>>>(x,  xbf);
    f32_to_bf16_kernel<<<(int)(wsz  / (8 * 256)), 256, 0, stream>>>(Wq, wqbf);
    f32_to_bf16_kernel<<<(int)(wsz  / (8 * 256)), 256, 0, stream>>>(Wk, wkbf);
    f32_to_bf16_kernel<<<(int)(wsz  / (8 * 256)), 256, 0, stream>>>(Wv, wvbf);
    f32_to_bf16_kernel<<<(int)(wsz  / (8 * 256)), 256, 0, stream>>>(Wo, wobf);

    gemm_bf16_wmma<true,  true ><<<grd, blk, 0, stream>>>(xbf, wqbf, qbf, Mrows, Dd, Dd);
    gemm_bf16_wmma<true,  true ><<<grd, blk, 0, stream>>>(xbf, wkbf, kbf, Mrows, Dd, Dd);
    gemm_bf16_wmma<true,  true ><<<grd, blk, 0, stream>>>(xbf, wvbf, vbf, Mrows, Dd, Dd);
    beta_kernel<<<(Mrows * Nh) / 256, 256, 0, stream>>>(x, Wb, bb, Dd, Nh);
    delta_rec<<<Bb * Nh, 32, 0, stream>>>(qbf, kbf, vbf, bb, attn, Ss, Nh, Hd, Dd);
    rmsnorm_bf16_kernel<<<Mrows, 256, 0, stream>>>(attn, nw, atbf, Dd);
    gemm_bf16_wmma<false, false><<<grd, blk, 0, stream>>>(atbf, wobf, (float*)d_out, Mrows, Dd, Dd);
}
